// SLiMPerformer_2680059592850
// MI455X (gfx1250) — matrix operand order repaired and checked by reading, not compiled, not run on hardware
//
#include <hip/hip_runtime.h>
#include <hip/hip_bf16.h>
#include <math.h>

// ---------------------------------------------------------------------------
// SLiM Performer forward on MI455X (gfx1250), wave32.
//  * All GEMMs on v_wmma_f32_16x16x32_f16 (f16 in HBM, f32 accum).
//  * GEMM tiles staged with GLOBAL_LOAD_ASYNC_TO_LDS_B128 (ASYNCcnt),
//    double-buffered: copy of tile k+1 overlaps WMMA on tile k.
//  * Weights stored transposed [N][K] f16 so B tiles are 16B-contiguous.
//  * Causal linear attention restructured as chunked prefix form (all WMMA).
// ---------------------------------------------------------------------------

#define L_    2
#define H_    8
#define VOC_  256
#define DM_   512
#define FF_   2048
#define B_    8
#define S_    8192
#define HD_   64
#define T_    1024
#define NC_   16      // chunks per sequence
#define CH_   64      // chunk length
#define MROWS (B_ * T_)   // 8192

typedef __attribute__((ext_vector_type(16))) _Float16 v16h;
typedef __attribute__((ext_vector_type(8)))  _Float16 v8h;
typedef __attribute__((ext_vector_type(8)))  float    v8f;

union HexV16 { v16h v; v8h h[2]; };

// ---- CDNA5 async global->LDS copy (ASYNCcnt) ------------------------------
__device__ inline unsigned lds_addr32(const void* p) {
  // generic LDS address: low 32 bits are the byte offset in the wave's LDS
  return (unsigned)(unsigned long long)p;
}
__device__ inline void async_b128(unsigned ldsoff, const _Float16* g) {
  asm volatile("global_load_async_to_lds_b128 %0, %1, off"
               :: "v"(ldsoff), "v"(g) : "memory");
}
__device__ inline void wait_async0() {
  asm volatile("s_wait_asynccnt 0x0" ::: "memory");
}

// A-fragment (16x32 f16, MxK): lane m = lane&15; K groups {0..7,16..23} for
// lanes 0-15 and {8..15,24..31} for lanes 16-31.  Two 16B LDS loads.
__device__ inline v16h ldsAfrag(const _Float16* tile, int ld, int lane) {
  const _Float16* p = tile + (lane & 15) * ld;
  int kb = (lane >> 4) << 3;                 // 0 or 8
  HexV16 u;
  u.h[0] = *(const v8h*)(p + kb);
  u.h[1] = *(const v8h*)(p + 16 + kb);
  return u.v;
}
// B-fragment (32x16 f16, KxN) from LDS stored transposed as [N][K]:
// lane n = lane&15; K groups {0..7,16..23} (lanes 0-15) / {8..15,24..31} (lanes 16-31).
__device__ inline v16h ldsBfrag(const _Float16* tile, int ld, int lane) {
  const _Float16* p = tile + (lane & 15) * ld;
  int kb = (lane >> 4) << 3;                 // 0 or 8
  HexV16 u;
  u.h[0] = *(const v8h*)(p + kb);
  u.h[1] = *(const v8h*)(p + 16 + kb);
  return u.v;
}
__device__ inline v8f wmma_f16(v16h a, v16h b, v8f c) {
  return __builtin_amdgcn_wmma_f32_16x16x32_f16(false, a, false, b,
                                                (short)0, c, false, false);
}

// ---------------------------------------------------------------------------
// Async double-buffered WMMA GEMM:
//   C[M,N] = epilogue(A[M,K](f16) @ Bt[N,K]^T(f16))
// 256 threads / 8 waves; BMxBN tile, BK=32 (one wmma K-step per stage).
// ---------------------------------------------------------------------------
enum { EPI_BIAS = 0, EPI_GELU = 1, EPI_RES = 2, EPI_FAVOR = 3 };

template<int BM, int BN, int EPI, typename CT>
__global__ __launch_bounds__(256) void gemm_wmma_kernel(
    const _Float16* __restrict__ A, long aBatch, int lda,
    const _Float16* __restrict__ Bt, long bBatch, int ldb,  // [N][K]
    const float* __restrict__ ep,          // bias[col]  or  rnorm[row*H+z]
    CT* __restrict__ C, long cBatch, int ldc,
    int K)
{
  constexpr int WM = 32;
  constexpr int WN = (BN == 128) ? 64 : 32;
  constexpr int TM = WM / 16, TN = WN / 16;
  constexpr int LD = 40;                 // 32 + 8 pad: 80B rows (16B multiple)
  __shared__ _Float16 As[2][BM * LD];
  __shared__ _Float16 Bs[2][BN * LD];

  const int tid  = threadIdx.x;
  const int wave = tid >> 5, lane = tid & 31;
  constexpr int WAVES_M = BM / WM;
  const int wm = (wave % WAVES_M) * WM;
  const int wn = (wave / WAVES_M) * WN;
  const int m0 = blockIdx.y * BM;
  const int n0 = blockIdx.x * BN;
  const _Float16* Ag = A  + (long)blockIdx.z * aBatch + (long)m0 * lda;
  const _Float16* Bg = Bt + (long)blockIdx.z * bBatch + (long)n0 * ldb;
  const long cOff = (long)blockIdx.z * cBatch;

  v8f acc[TM][TN];
  #pragma unroll
  for (int i = 0; i < TM; ++i)
    #pragma unroll
    for (int j = 0; j < TN; ++j) { v8f z = {}; acc[i][j] = z; }

  // one K-stage = (BM+BN) rows x 64B, issued as per-lane b128 async copies
  auto stage = [&](int buf, int k0) {
    #pragma unroll
    for (int it = 0; it < (BM * 4) / 256; ++it) {
      int lin = tid + it * 256;
      int r = lin >> 2, ck = (lin & 3) << 3;
      async_b128(lds_addr32(&As[buf][r * LD + ck]),
                 Ag + (long)r * lda + k0 + ck);
    }
    #pragma unroll
    for (int it = 0; it < (BN * 4) / 256; ++it) {
      int lin = tid + it * 256;
      int r = lin >> 2, ck = (lin & 3) << 3;
      async_b128(lds_addr32(&Bs[buf][r * LD + ck]),
                 Bg + (long)r * ldb + k0 + ck);
    }
  };

  stage(0, 0);
  const int nk = K >> 5;
  for (int i = 0; i < nk; ++i) {
    const int cur = i & 1;
    wait_async0();            // this wave's copies for tile i are in LDS
    __syncthreads();          // everyone's copies landed; prev buf free
    if (i + 1 < nk) stage(cur ^ 1, (i + 1) << 5);

    v16h bfrag[TN];
    #pragma unroll
    for (int j = 0; j < TN; ++j)
      bfrag[j] = ldsBfrag(&Bs[cur][(wn + j * 16) * LD], LD, lane);
    #pragma unroll
    for (int i2 = 0; i2 < TM; ++i2) {
      v16h afrag = ldsAfrag(&As[cur][(wm + i2 * 16) * LD], LD, lane);
      #pragma unroll
      for (int j = 0; j < TN; ++j)
        acc[i2][j] = wmma_f16(afrag, bfrag[j], acc[i2][j]);
    }
  }

  const int hi = lane >> 4, nl = lane & 15;
  #pragma unroll
  for (int i = 0; i < TM; ++i)
    #pragma unroll
    for (int j = 0; j < TN; ++j)
      #pragma unroll
      for (int r = 0; r < 8; ++r) {
        int grow = m0 + wm + i * 16 + r + 8 * hi;
        int gcol = n0 + wn + j * 16 + nl;
        float a = acc[i][j][r];
        CT* cp = C + cOff + (long)grow * ldc + gcol;
        if constexpr (EPI == EPI_RES) {
          *cp = (CT)((float)*cp + a + ep[gcol]);
        } else {
          float val;
          if constexpr (EPI == EPI_BIAS) {
            val = a + ep[gcol];
          } else if constexpr (EPI == EPI_GELU) {
            float xg = a + ep[gcol];
            val = 0.5f * xg * (1.0f + erff(xg * 0.70710678118654752f));
          } else { // EPI_FAVOR: exp(acc - 0.5*|zs|^2), rnorm pre-scaled
            val = expf(a - ep[(long)grow * H_ + (int)blockIdx.z]);
          }
          *cp = (CT)val;
        }
      }
}

// ---------------------------------------------------------------------------
// Chunked linear attention, stage 1:  S[b,h,c] = kf_chunk^T @ v_chunk (64x64)
//                                     z[b,h,c][m] = sum_t kf[t][m]
// ---------------------------------------------------------------------------
__global__ __launch_bounds__(128) void chunk_kv_kernel(
    const _Float16* __restrict__ kf, const _Float16* __restrict__ v,
    float* __restrict__ S, float* __restrict__ z)
{
  constexpr int LD = 72;                       // 64 + 8 pad
  __shared__ _Float16 KT[64 * LD];             // [m][t]
  __shared__ _Float16 VT[64 * LD];             // [d][t]
  const int tid = threadIdx.x;
  const int c = blockIdx.x, h = blockIdx.y, b = blockIdx.z;
  const long rowBase = (long)(b * T_ + c * CH_) * DM_ + h * HD_;

  #pragma unroll 4
  for (int it = 0; it < 32; ++it) {
    int lin = tid + it * 128;                  // 0..4095
    int t = lin >> 6, m = lin & 63;
    KT[m * LD + t] = kf[rowBase + (long)t * DM_ + m];
    VT[m * LD + t] = v [rowBase + (long)t * DM_ + m];
  }
  __syncthreads();

  const int wave = tid >> 5, lane = tid & 31;
  const int wm = wave * 16;
  v8f acc[4];
  #pragma unroll
  for (int j = 0; j < 4; ++j) { v8f zz = {}; acc[j] = zz; }
  #pragma unroll
  for (int ks = 0; ks < 2; ++ks) {
    v16h a = ldsAfrag(KT + wm * LD + ks * 32, LD, lane);
    #pragma unroll
    for (int j = 0; j < 4; ++j) {
      v16h bf = ldsBfrag(VT + (j * 16) * LD + ks * 32, LD, lane);
      acc[j] = wmma_f16(a, bf, acc[j]);
    }
  }
  const long sBase = (((long)(b * H_ + h)) * NC_ + c) * 4096;
  const int hi = lane >> 4, nl = lane & 15;
  #pragma unroll
  for (int j = 0; j < 4; ++j)
    #pragma unroll
    for (int r = 0; r < 8; ++r) {
      int m = wm + r + 8 * hi, d = j * 16 + nl;
      S[sBase + m * 64 + d] = acc[j][r];
    }
  if (tid < 64) {
    float s = 0.f;
    for (int t = 0; t < 64; ++t) s += (float)KT[tid * LD + t];
    z[(((long)(b * H_ + h)) * NC_ + c) * 64 + tid] = s;
  }
}

// Exclusive prefix over the 16 chunks (per b,h): state and normalizer.
__global__ void prefix_state_kernel(const float* __restrict__ S,
                                    float* __restrict__ P) {
  int idx = blockIdx.x * 256 + threadIdx.x;    // < 64 * 4096
  int bh = idx >> 12, e = idx & 4095;
  float acc = 0.f;
  for (int c = 0; c < NC_; ++c) {
    long o = (((long)bh) * NC_ + c) * 4096 + e;
    P[o] = acc; acc += S[o];
  }
}
__global__ void prefix_z_kernel(const float* __restrict__ z,
                                float* __restrict__ zP) {
  int idx = blockIdx.x * 256 + threadIdx.x;    // < 64 * 64
  int bh = idx >> 6, m = idx & 63;
  float acc = 0.f;
  for (int c = 0; c < NC_; ++c) {
    long o = (((long)bh) * NC_ + c) * 64 + m;
    zP[o] = acc; acc += z[o];
  }
}

// ---------------------------------------------------------------------------
// Attention stage 2 (fused):  per (b,h,chunk)
//   T1  = qf @ kf^T            (causal mask tk<=tq)
//   num = [qf | T1m] @ [P ; v]      (K = 128)
//   den = rowsum(T1m) + qf . zP
//   h  += num / (den + 1e-16)
// ---------------------------------------------------------------------------
__global__ __launch_bounds__(128) void attn_kernel(
    const _Float16* __restrict__ qf, const _Float16* __restrict__ kf,
    const _Float16* __restrict__ v,  const float* __restrict__ P,
    const float* __restrict__ zP, float* __restrict__ hout)
{
  constexpr int LD = 72, LD2 = 136;            // padded, 16B-aligned rows
  __shared__ _Float16 Qs [64 * LD];            // [tq][feat]
  __shared__ _Float16 Ks [64 * LD];            // [tk][feat]
  __shared__ _Float16 A2 [64 * LD2];           // [tq][0..63]=qf, [64..127]=T1m
  __shared__ _Float16 VT2[64 * LD2];           // [d][k]: k<64 -> P, k>=64 -> v
  __shared__ float zPl[64];
  __shared__ float denl[64];

  const int tid = threadIdx.x;
  const int c = blockIdx.x, h = blockIdx.y, b = blockIdx.z;
  const long rowBase = (long)(b * T_ + c * CH_) * DM_ + h * HD_;
  const long pBase   = (((long)(b * H_ + h)) * NC_ + c) * 4096;

  #pragma unroll 4
  for (int it = 0; it < 32; ++it) {
    int lin = tid + it * 128;
    int t = lin >> 6, f = lin & 63;
    _Float16 qv = qf[rowBase + (long)t * DM_ + f];
    Qs[t * LD + f]  = qv;
    A2[t * LD2 + f] = qv;
    Ks[t * LD + f]  = kf[rowBase + (long)t * DM_ + f];
    VT2[f * LD2 + t]      = (_Float16)P[pBase + t * 64 + f];   // P[m=t][d=f]
    VT2[f * LD2 + 64 + t] = v[rowBase + (long)t * DM_ + f];
  }
  if (tid < 64) zPl[tid] = zP[(((long)(b * H_ + h)) * NC_ + c) * 64 + tid];
  __syncthreads();

  const int wave = tid >> 5, lane = tid & 31;
  const int wm = wave * 16;
  const int hi = lane >> 4, nl = lane & 15;

  { // phase 1: T1 = Qs @ Ks^T, masked into A2[:, 64..127]
    v8f acc[4];
    #pragma unroll
    for (int j = 0; j < 4; ++j) { v8f zz = {}; acc[j] = zz; }
    #pragma unroll
    for (int ks = 0; ks < 2; ++ks) {
      v16h a = ldsAfrag(Qs + wm * LD + ks * 32, LD, lane);
      #pragma unroll
      for (int j = 0; j < 4; ++j) {
        v16h bf = ldsBfrag(Ks + (j * 16) * LD + ks * 32, LD, lane);
        acc[j] = wmma_f16(a, bf, acc[j]);
      }
    }
    #pragma unroll
    for (int j = 0; j < 4; ++j)
      #pragma unroll
      for (int r = 0; r < 8; ++r) {
        int tq = wm + r + 8 * hi, tk = j * 16 + nl;
        float x = (tk <= tq) ? acc[j][r] : 0.f;
        A2[tq * LD2 + 64 + tk] = (_Float16)x;
      }
  }
  __syncthreads();

  if (tid < 64) { // denominator
    float s = 0.f;
    for (int tk = 0; tk < 64; ++tk) s += (float)A2[tid * LD2 + 64 + tk];
    for (int m = 0; m < 64; ++m)   s += (float)Qs[tid * LD + m] * zPl[m];
    denl[tid] = s;
  }
  __syncthreads();

  // phase 2: num = A2 @ [P ; v]  (K = 128)
  v8f acc[4];
  #pragma unroll
  for (int j = 0; j < 4; ++j) { v8f zz = {}; acc[j] = zz; }
  #pragma unroll
  for (int ks = 0; ks < 4; ++ks) {
    v16h a = ldsAfrag(A2 + wm * LD2 + ks * 32, LD2, lane);
    #pragma unroll
    for (int j = 0; j < 4; ++j) {
      v16h bf = ldsBfrag(VT2 + (j * 16) * LD2 + ks * 32, LD2, lane);
      acc[j] = wmma_f16(a, bf, acc[j]);
    }
  }
  #pragma unroll
  for (int j = 0; j < 4; ++j)
    #pragma unroll
    for (int r = 0; r < 8; ++r) {
      int tq = wm + r + 8 * hi, d = j * 16 + nl;
      float o = acc[j][r] / (denl[tq] + 1e-16f);
      hout[rowBase + (long)tq * DM_ + d] += o;    // residual add
    }
}

// ---------------------------------------------------------------------------
// Support kernels
// ---------------------------------------------------------------------------
__global__ __launch_bounds__(256) void ln_kernel(
    const float* __restrict__ x, const float* __restrict__ g,
    const float* __restrict__ be, _Float16* __restrict__ y)
{
  const int row  = blockIdx.x * 8 + (threadIdx.x >> 5);
  const int lane = threadIdx.x & 31;
  const float* xr = x + (long)row * DM_;
  float vbuf[16]; float s = 0.f, s2 = 0.f;
  #pragma unroll
  for (int i = 0; i < 16; ++i) {
    float t = xr[lane + i * 32]; vbuf[i] = t; s += t; s2 += t * t;
  }
  #pragma unroll
  for (int o = 16; o >= 1; o >>= 1) {
    s  += __shfl_xor(s,  o, 32);
    s2 += __shfl_xor(s2, o, 32);
  }
  float mean = s * (1.f / DM_);
  float var  = s2 * (1.f / DM_) - mean * mean;
  float rs   = rsqrtf(var + 1e-5f);
  _Float16* yr = y + (long)row * DM_;
  #pragma unroll
  for (int i = 0; i < 16; ++i) {
    int cc = lane + i * 32;
    yr[cc] = (_Float16)((vbuf[i] - mean) * rs * g[cc] + be[cc]);
  }
}

// rnorm[row,h] = 0.5 * HD^-0.5 * |q_head|^2   (zs scale folded: 0.5/8)
__global__ void rnorm_kernel(const _Float16* __restrict__ q,
                             float* __restrict__ rn) {
  int idx = blockIdx.x * 256 + threadIdx.x;     // < MROWS * H_
  int row = idx >> 3, hh = idx & 7;
  const _Float16* p = q + (long)row * DM_ + hh * HD_;
  float s = 0.f;
  for (int i = 0; i < HD_; ++i) { float t = (float)p[i]; s += t * t; }
  rn[idx] = 0.0625f * s;
}

__global__ void cvt_kernel(const float* __restrict__ src,
                           _Float16* __restrict__ dst, float scale, int n) {
  int i = blockIdx.x * 256 + threadIdx.x;
  if (i < n) dst[i] = (_Float16)(src[i] * scale);
}

// batched f32[K,N] -> f16 transposed [N,K]
__global__ void cvtT_kernel(const float* __restrict__ src,
                            _Float16* __restrict__ dst,
                            int K, int N, float scale, int total) {
  int idx = blockIdx.x * 256 + threadIdx.x;
  if (idx >= total) return;
  int bi  = idx / (K * N);
  int rem = idx - bi * (K * N);
  int k = rem / N, n = rem - k * N;
  dst[(long)bi * K * N + (long)n * K + k] = (_Float16)(src[idx] * scale);
}

// token embed + sin/cos positions, reshaped [B, T, DM]
__global__ void embed_kernel(const int* __restrict__ x,
                             const float* __restrict__ emb,
                             float* __restrict__ h) {
  int idx = blockIdx.x * 256 + threadIdx.x;     // < MROWS * DM_
  int col = idx & 511;
  int rowbt = idx >> 9;
  int b = rowbt >> 10, t = rowbt & 1023;
  int sc = col >> 6, j = col & 63;
  int s = t * 8 + sc;
  float val;
  const float NEG_LOG1E4_OVER_HALFVD = -9.210340371976184f / 32.f;
  if (j < 32) {
    int tok = x[b * S_ + s];
    val = emb[tok * 32 + j];
  } else if (j < 48) {
    float fr = expf(NEG_LOG1E4_OVER_HALFVD * (float)(2 * (j - 32)));
    val = sinf((float)s * fr);
  } else {
    float fr = expf(NEG_LOG1E4_OVER_HALFVD * (float)(2 * (j - 48)));
    val = cosf((float)s * fr);
  }
  h[idx] = val;
}

// ---------------------------------------------------------------------------
extern "C" void kernel_launch(void* const* d_in, const int* in_sizes, int n_in,
                              void* d_out, int out_size, void* d_ws, size_t ws_size,
                              hipStream_t stream) {
  (void)in_sizes; (void)n_in; (void)out_size; (void)ws_size;
  const int*   x    = (const int*)  d_in[0];
  const float* rfs  = (const float*)d_in[1];
  const float* emb  = (const float*)d_in[2];
  const float* Wq   = (const float*)d_in[3];
  const float* bq   = (const float*)d_in[4];
  const float* Wk   = (const float*)d_in[5];
  const float* bk   = (const float*)d_in[6];
  const float* Wv   = (const float*)d_in[7];
  const float* bv   = (const float*)d_in[8];
  const float* U    = (const float*)d_in[9];
  const float* bU   = (const float*)d_in[10];
  const float* Vw   = (const float*)d_in[11];
  const float* bV   = (const float*)d_in[12];
  const float* g1   = (const float*)d_in[13];
  const float* be1  = (const float*)d_in[14];
  const float* g2   = (const float*)d_in[15];
  const float* be2  = (const float*)d_in[16];
  const float* Wout = (const float*)d_in[17];
  const float* bout = (const float*)d_in[18];
  float* out = (float*)d_out;

  // workspace arena
  char* ws = (char*)d_ws;
  size_t off = 0;
  auto alloc = [&](size_t bytes) -> char* {
    char* p = ws + off; off += (bytes + 255) & ~(size_t)255; return p;
  };
  float*    h    = (float*)   alloc((size_t)MROWS * DM_ * 4);
  _Float16* hn16 = (_Float16*)alloc((size_t)MROWS * DM_ * 2);
  _Float16* q16  = (_Float16*)alloc((size_t)MROWS * DM_ * 2);
  _Float16* k16  = (_Float16*)alloc((size_t)MROWS * DM_ * 2);
  _Float16* v16  = (_Float16*)alloc((size_t)MROWS * DM_ * 2);
  _Float16* ff16 = (_Float16*)alloc((size_t)MROWS * FF_ * 2);
  _Float16* qf16 = ff16;                         // alias: attention phase only
  _Float16* kf16 = ff16 + (size_t)MROWS * DM_;   // alias: attention phase only
  float* rq  = (float*)alloc((size_t)MROWS * H_ * 4);
  float* rk  = (float*)alloc((size_t)MROWS * H_ * 4);
  float* Sst = (float*)alloc((size_t)B_ * H_ * NC_ * 4096 * 4);
  float* Pst = (float*)alloc((size_t)B_ * H_ * NC_ * 4096 * 4);
  float* zst = (float*)alloc((size_t)B_ * H_ * NC_ * 64 * 4);
  float* zPs = (float*)alloc((size_t)B_ * H_ * NC_ * 64 * 4);
  _Float16* WqT   = (_Float16*)alloc((size_t)L_ * DM_ * DM_ * 2);
  _Float16* WkT   = (_Float16*)alloc((size_t)L_ * DM_ * DM_ * 2);
  _Float16* WvT   = (_Float16*)alloc((size_t)L_ * DM_ * DM_ * 2);
  _Float16* UT    = (_Float16*)alloc((size_t)L_ * DM_ * FF_ * 2);
  _Float16* VwT   = (_Float16*)alloc((size_t)L_ * FF_ * DM_ * 2);
  _Float16* rfsT  = (_Float16*)alloc((size_t)L_ * H_ * HD_ * HD_ * 2);
  _Float16* WoutT = (_Float16*)alloc((size_t)DM_ * VOC_ * 2);

  auto cvtT = [&](const float* s, _Float16* d, int K, int N, int batch, float sc) {
    int total = batch * K * N;
    cvtT_kernel<<<dim3((total + 255) / 256), dim3(256), 0, stream>>>(s, d, K, N, sc, total);
  };
  cvtT(Wq,   WqT,   DM_, DM_, L_,       1.f);
  cvtT(Wk,   WkT,   DM_, DM_, L_,       1.f);
  cvtT(Wv,   WvT,   DM_, DM_, L_,       1.f);
  cvtT(U,    UT,    DM_, FF_, L_,       1.f);
  cvtT(Vw,   VwT,   FF_, DM_, L_,       1.f);
  cvtT(rfs,  rfsT,  HD_, HD_, L_ * H_,  0.35355339059327373f);  // HD^-0.25
  cvtT(Wout, WoutT, DM_, VOC_, 1,       1.f);

  embed_kernel<<<dim3((MROWS * DM_) / 256), dim3(256), 0, stream>>>(x, emb, h);

  for (int l = 0; l < L_; ++l) {
    const _Float16* wq = WqT + (size_t)l * DM_ * DM_;
    const _Float16* wk = WkT + (size_t)l * DM_ * DM_;
    const _Float16* wv = WvT + (size_t)l * DM_ * DM_;
    const _Float16* wu = UT  + (size_t)l * DM_ * FF_;
    const _Float16* wV = VwT + (size_t)l * FF_ * DM_;
    const _Float16* wr = rfsT + (size_t)l * H_ * HD_ * HD_;

    // ---- attention block ----
    ln_kernel<<<dim3(MROWS / 8), dim3(256), 0, stream>>>(h, g1 + l * DM_, be1 + l * DM_, hn16);
    gemm_wmma_kernel<128, 128, EPI_BIAS, _Float16>
        <<<dim3(DM_ / 128, MROWS / 128, 1), 256, 0, stream>>>(
        hn16, 0, DM_, wq, 0, DM_, bq + l * DM_, q16, 0, DM_, DM_);
    gemm_wmma_kernel<128, 128, EPI_BIAS, _Float16>
        <<<dim3(DM_ / 128, MROWS / 128, 1), 256, 0, stream>>>(
        hn16, 0, DM_, wk, 0, DM_, bk + l * DM_, k16, 0, DM_, DM_);
    gemm_wmma_kernel<128, 128, EPI_BIAS, _Float16>
        <<<dim3(DM_ / 128, MROWS / 128, 1), 256, 0, stream>>>(
        hn16, 0, DM_, wv, 0, DM_, bv + l * DM_, v16, 0, DM_, DM_);

    rnorm_kernel<<<dim3((MROWS * H_) / 256), dim3(256), 0, stream>>>(q16, rq);
    rnorm_kernel<<<dim3((MROWS * H_) / 256), dim3(256), 0, stream>>>(k16, rk);

    // FAVOR features per head (grid.z = head)
    gemm_wmma_kernel<128, 64, EPI_FAVOR, _Float16>
        <<<dim3(1, MROWS / 128, H_), 256, 0, stream>>>(
        q16, 64, DM_, wr, HD_ * HD_, HD_, rq, qf16, 64, DM_, HD_);
    gemm_wmma_kernel<128, 64, EPI_FAVOR, _Float16>
        <<<dim3(1, MROWS / 128, H_), 256, 0, stream>>>(
        k16, 64, DM_, wr, HD_ * HD_, HD_, rk, kf16, 64, DM_, HD_);

    chunk_kv_kernel<<<dim3(NC_, H_, B_), dim3(128), 0, stream>>>(kf16, v16, Sst, zst);
    prefix_state_kernel<<<dim3((B_ * H_ * 4096) / 256), dim3(256), 0, stream>>>(Sst, Pst);
    prefix_z_kernel<<<dim3((B_ * H_ * 64) / 256), dim3(256), 0, stream>>>(zst, zPs);
    attn_kernel<<<dim3(NC_, H_, B_), dim3(128), 0, stream>>>(qf16, kf16, v16, Pst, zPs, h);

    // ---- FFN applied twice with shared weights ----
    for (int rep = 0; rep < 2; ++rep) {
      ln_kernel<<<dim3(MROWS / 8), dim3(256), 0, stream>>>(h, g2 + l * DM_, be2 + l * DM_, hn16);
      gemm_wmma_kernel<128, 128, EPI_GELU, _Float16>
          <<<dim3(FF_ / 128, MROWS / 128, 1), 256, 0, stream>>>(
          hn16, 0, DM_, wu, 0, DM_, bU + l * FF_, ff16, 0, FF_, DM_);
      gemm_wmma_kernel<128, 128, EPI_RES, float>
          <<<dim3(DM_ / 128, MROWS / 128, 1), 256, 0, stream>>>(
          ff16, 0, FF_, wV, 0, FF_, bV + l * DM_, h, 0, DM_, FF_);
    }
  }

  // final projection: out = h @ Wout + bout
  cvt_kernel<<<dim3((MROWS * DM_) / 256), dim3(256), 0, stream>>>(h, hn16, 1.f, MROWS * DM_);
  gemm_wmma_kernel<128, 128, EPI_BIAS, float>
      <<<dim3(VOC_ / 128, MROWS / 128, 1), 256, 0, stream>>>(
      hn16, 0, DM_, WoutT, 0, DM_, bout, out, 0, VOC_, DM_);
}
